// differentiableSuperpixelEmbedding_68169720922170
// MI455X (gfx1250) — matrix operand order repaired and checked by reading, not compiled, not run on hardware
//
#include <hip/hip_runtime.h>
#include <hip/hip_fp16.h>

typedef __attribute__((ext_vector_type(16))) _Float16 v16h;
typedef __attribute__((ext_vector_type(8)))  float    v8f;

#define B_ 8
#define S_ 64
#define C_ 3
#define H_ 224
#define W_ 224
#define E_ 128
#define HW_ (H_ * W_)            // 50176
#define TILES_PER_IMG (HW_ / 16) // 3136 (224 = 14*16 -> tiles never split rows)
#define PATCHES_ 196             // 14*14

static __device__ __forceinline__ unsigned pack2h(_Float16 lo, _Float16 hi) {
    union { _Float16 h[2]; unsigned u; } p;
    p.h[0] = lo; p.h[1] = hi;
    return p.u;
}

// ---------------------------------------------------------------------------
// Kernel 1: 3x3 spixel conv (SAME) + argmax over 64 channels -> labels.
// GEMM form: [16 pixels x K=27(pad32)] x [27 x 64] via v_wmma_f32_16x16x32_f16.
// One wave = 16 pixels; 4 WMMAs cover all 64 segment channels, full K each.
//  - B fragments (lane-only dependent, identical for every wave/block) are
//    built ONCE per block into LDS in the exact WMMA register layout, so the
//    per-wave path is just 2x ds_load_b128 per tile.
//  - A tap loads are branchless (clamped address + select).
//  - Argmax is branchless: monotonic u32 key | (63-idx) packed in a u64 max.
// ---------------------------------------------------------------------------
__global__ __launch_bounds__(256) void k_spixel_argmax(
    const float* __restrict__ img, const float* __restrict__ w_spixel,
    const float* __restrict__ b_spixel, int* __restrict__ labels)
{
    // B staging: [tile][lane][dword] packed f16 pairs, WMMA B layout:
    // lane holds column N = 16*tile + lane%16; dword d holds K = 2d+16*(lane/16)
    // and K+1 (K >= 27 zero-padded).  w_spixel is [S,27] row-major.
    __shared__ __align__(32) unsigned bmat[4 * 32 * 8];   // 4 KB

    const int tid = threadIdx.x;
    for (int idx = tid; idx < 4 * 32 * 8; idx += 256) {
        const int t = idx >> 8, l = (idx >> 3) & 31, d = idx & 7;
        const int n  = 16 * t + (l & 15);
        const int K0 = 2 * d + 16 * (l >> 4);
        const int K1 = K0 + 1;
        const float v0 = w_spixel[n * 27 + (K0 < 27 ? K0 : 0)];
        const float v1 = w_spixel[n * 27 + (K1 < 27 ? K1 : 0)];
        bmat[idx] = pack2h(K0 < 27 ? (_Float16)v0 : (_Float16)0.f,
                           K1 < 27 ? (_Float16)v1 : (_Float16)0.f);
    }
    __syncthreads();

    const int lane = tid & 31;
    const int half = lane >> 4;           // 0: lanes 0-15, 1: lanes 16-31
    const int nloc = lane & 15;
    const int waveGlobal = blockIdx.x * 8 + (tid >> 5);
    const int b    = waveGlobal / TILES_PER_IMG;
    const int tile = waveGlobal % TILES_PER_IMG;

    const int pix = tile * 16 + nloc;     // this lane's A-row pixel
    const int h = pix / W_;
    const int w = pix % W_;
    const float* imgb = img + (size_t)b * C_ * HW_;

    // A-matrix (16x32 f16): lane holds row M=lane%16; element j holds
    // K = j + 16*half   (contiguous order, matching B staging).
    // K index k = c*9 + (dy+1)*3 + (dx+1); K>=27 and OOB taps are zero.
    v16h a;
#pragma unroll
    for (int j = 0; j < 16; ++j) {
        const int K  = j + 16 * half;
        const int Kc = (K < 27) ? K : 0;          // safe address
        const int c = Kc / 9, rem = Kc % 9;
        const int hh = h + rem / 3 - 1, ww = w + rem % 3 - 1;
        const bool valid =
            (K < 27) & (hh >= 0) & (hh < H_) & (ww >= 0) & (ww < W_);
        const int hcl = hh < 0 ? 0 : (hh > H_ - 1 ? H_ - 1 : hh);
        const int wcl = ww < 0 ? 0 : (ww > W_ - 1 ? W_ - 1 : ww);
        const float x = imgb[c * HW_ + hcl * W_ + wcl];   // unconditional load
        a[j] = valid ? (_Float16)x : (_Float16)0.f;       // select, no branch
    }

    // 4 WMMAs: segment tiles t -> columns 16t..16t+15; B from LDS.
    v8f acc[4];
#pragma unroll
    for (int t = 0; t < 4; ++t) {
        const v16h bt =
            *(const v16h*)&bmat[(t * 32 + lane) * 8];     // 2x ds_load_b128
        v8f cz = {};
        acc[t] = __builtin_amdgcn_wmma_f32_16x16x32_f16(
            false, a, false, bt, (short)0, cz, false, false);
    }

    // Branchless argmax over the 64 segments.  D layout: VGPR r of lane holds
    // D[M = r + 8*half, N = lane%16].  Pack (monotonic-key(logit) << 32) |
    // (63 - seg) and max-reduce across the 16 lanes of the half-group.
#pragma unroll
    for (int r = 0; r < 8; ++r) {
        unsigned long long best = 0ull;
#pragma unroll
        for (int t = 0; t < 4; ++t) {
            const float v = acc[t][r] + b_spixel[16 * t + nloc];
            const unsigned u = __float_as_uint(v);
            const unsigned key =
                (u & 0x80000000u) ? ~u : (u | 0x80000000u); // order-preserving
            const unsigned long long k =
                ((unsigned long long)key << 32) |
                (unsigned)(63 - (16 * t + nloc));           // low idx wins ties
            best = (t == 0) ? k : (k > best ? k : best);
        }
#pragma unroll
        for (int msk = 1; msk < 16; msk <<= 1) {
            const unsigned hi = __shfl_xor((unsigned)(best >> 32), msk, 32);
            const unsigned lo = __shfl_xor((unsigned)best, msk, 32);
            const unsigned long long o = ((unsigned long long)hi << 32) | lo;
            best = (o > best) ? o : best;
        }
        if (nloc == 0)
            labels[(size_t)b * HW_ + tile * 16 + (r + 8 * half)] =
                63 - (int)(best & 0xFFu);
    }
}

// ---------------------------------------------------------------------------
// Kernel 2: per 16x16 patch, project each pixel (3ch -> 128) with the patch
// weight at its within-patch position and scatter-accumulate by label into
// LDS acc[64][128] (ds_add_f32), then flush present rows with
// global_atomic_add_f32 into g_acc[b,s,e].
// ---------------------------------------------------------------------------
__global__ __launch_bounds__(256) void k_scatter(
    const float* __restrict__ img, const float* __restrict__ w_patch,
    const int* __restrict__ labels, float* __restrict__ g_acc,
    unsigned int* __restrict__ g_present)
{
    __shared__ float acc[S_ * E_];          // 32 KB
    __shared__ unsigned int present[S_];

    const int tid = threadIdx.x;
    const int b     = blockIdx.x / PATCHES_;
    const int patch = blockIdx.x % PATCHES_;
    const int py = patch / 14, px = patch % 14;

    for (int i = tid; i < S_ * E_; i += 256) acc[i] = 0.f;
    if (tid < S_) present[tid] = 0u;
    __syncthreads();

    const int i = tid >> 4, j = tid & 15;               // within-patch pos
    const int h = py * 16 + i, w0 = px * 16 + j;
    const size_t poff = (size_t)b * C_ * HW_ + (size_t)h * W_ + w0;
    const float x0 = img[poff];
    const float x1 = img[poff + HW_];
    const float x2 = img[poff + 2 * HW_];
    const int l = labels[(size_t)b * HW_ + (size_t)h * W_ + w0];
    atomicOr(&present[l], 1u);

    float* row = &acc[l * E_];
    const float* wp = w_patch + (i * 16 + j);           // [E,C,16,16]
#pragma unroll 4
    for (int e = 0; e < E_; ++e) {
        const float* we = wp + (size_t)e * (C_ * 256);
        float v = x0 * we[0] + x1 * we[256] + x2 * we[512];
        atomicAdd(&row[e], v);                          // ds_add_f32
    }
    __syncthreads();

    for (int idx = tid; idx < S_ * E_; idx += 256) {
        int s = idx >> 7;
        if (present[s])
            unsafeAtomicAdd(&g_acc[((size_t)b * S_ + s) * E_ + (idx & 127)],
                            acc[idx]);                  // global_atomic_add_f32
    }
    if (tid < S_ && present[tid]) atomicOr(&g_present[b * S_ + tid], 1u);
}

// ---------------------------------------------------------------------------
// Kernel 3: build uniq (sorted present labels, zero-padded -- exact
// torch.unique-replication semantics of the reference) and emit
// out[b,s,e] = b_patch[e] + g_acc[b, uniq[s], e] / 196.
// ---------------------------------------------------------------------------
__global__ __launch_bounds__(256) void k_gather(
    const float* __restrict__ g_acc, const unsigned int* __restrict__ g_present,
    const float* __restrict__ b_patch, float* __restrict__ out)
{
    __shared__ int uniq[S_];
    const int b = blockIdx.x;
    const int tid = threadIdx.x;
    if (tid == 0) {
        int cnt = 0;
        for (int s = 0; s < S_; ++s)
            if (g_present[b * S_ + s]) uniq[cnt++] = s;
        for (; cnt < S_; ++cnt) uniq[cnt] = 0;
    }
    __syncthreads();
    const float inv = 1.f / 196.f;
    for (int idx = tid; idx < S_ * E_; idx += 256) {
        int s = idx >> 7, e = idx & 127;
        int l = uniq[s];
        out[((size_t)b * S_ + s) * E_ + e] =
            b_patch[e] + g_acc[((size_t)b * S_ + l) * E_ + e] * inv;
    }
}

// ---------------------------------------------------------------------------
extern "C" void kernel_launch(void* const* d_in, const int* in_sizes, int n_in,
                              void* d_out, int out_size, void* d_ws, size_t ws_size,
                              hipStream_t stream)
{
    (void)in_sizes; (void)n_in; (void)out_size; (void)ws_size;
    const float* img      = (const float*)d_in[0];
    const float* w_spixel = (const float*)d_in[1];
    const float* b_spixel = (const float*)d_in[2];
    const float* w_patch  = (const float*)d_in[3];
    const float* b_patch  = (const float*)d_in[4];
    float* out = (float*)d_out;

    // Workspace layout: labels (B*HW int) | g_acc (B*S*E f32) | g_present (B*S u32)
    char* ws = (char*)d_ws;
    int* labels = (int*)ws;
    float* g_acc = (float*)(ws + sizeof(int) * (size_t)B_ * HW_);
    unsigned int* g_present =
        (unsigned int*)(ws + sizeof(int) * (size_t)B_ * HW_ +
                        sizeof(float) * (size_t)B_ * S_ * E_);

    // Zero accumulators + presence flags (contiguous) — capture-safe memset.
    hipMemsetAsync(g_acc, 0,
                   sizeof(float) * (size_t)B_ * S_ * E_ +
                   sizeof(unsigned int) * (size_t)B_ * S_, stream);

    k_spixel_argmax<<<TILES_PER_IMG * B_ / 8, 256, 0, stream>>>(
        img, w_spixel, b_spixel, labels);
    k_scatter<<<B_ * PATCHES_, 256, 0, stream>>>(
        img, w_patch, labels, g_acc, g_present);
    k_gather<<<B_, 256, 0, stream>>>(g_acc, g_present, b_patch, out);
}
